// MultiHeadLatentAttention_11897059410589
// MI455X (gfx1250) — matrix operand order repaired and checked by reading, not compiled, not run on hardware
//
#include <hip/hip_runtime.h>
#include <math.h>

// ---------------------------------------------------------------------------
// MLA (Multi-Head Latent Attention) for MI455X / gfx1250.
// B=4, T=2048, C=768, H=12, D=64, R=64. All matmuls via v_wmma_f32_16x16x32_bf16.
// GEMMs: 32x64 tile per wave (2 A-frags share each B-frag -> 1.5 loads/WMMA).
// Attention: flash with online softmax, K-tile prefetch + early V loads.
// ---------------------------------------------------------------------------

typedef __attribute__((ext_vector_type(16))) __bf16 v16bf;
typedef __attribute__((ext_vector_type(8)))  __bf16 v8bf;
typedef __attribute__((ext_vector_type(8)))  float  v8f;

#define Bc   4
#define Tc   2048
#define Cc   768
#define Hc   12
#define Dc   64
#define Rc   64
#define BTc  (Bc * Tc)          // 8192 rows
#define LOG_THETA 9.210340371976184f   // ln(10000)

union F8  { v8f v; float f[8]; };
union BF16x16 { v16bf v; v8bf h[2]; };

__device__ __forceinline__ __bf16 f2bf(float x) {
  union { float f; unsigned u; } a; a.f = x;
  unsigned r = a.u + 0x7FFFu + ((a.u >> 16) & 1u);   // round-to-nearest-even
  unsigned short hs = (unsigned short)(r >> 16);
  union { unsigned short s; __bf16 b; } c; c.s = hs; return c.b;
}

// A fragment: 16x32 bf16, ISA layout: lane m=L&15, half=L>>4;
// elems 0..7 -> k = half*8 + e ; elems 8..15 -> k = 16 + half*8 + (e-8)
__device__ __forceinline__ v16bf load_a_frag(const __bf16* __restrict__ A, int lda, int kb) {
  int lane = threadIdx.x & 31;
  int m = lane & 15, half = lane >> 4;
  const __bf16* p = A + (size_t)m * lda + kb + half * 8;
  BF16x16 u;
  u.h[0] = *(const v8bf*)p;
  u.h[1] = *(const v8bf*)(p + 16);
  return u.v;
}

// B fragment from transposed operand Bt[N][K]: lane n = nb + (L&15),
// elems 0..7 -> k = kb + (L>>4)*8 + e ; elems 8..15 -> k = kb + 16 + (L>>4)*8 + (e-8)
__device__ __forceinline__ v16bf load_b_frag(const __bf16* __restrict__ Bt, int ldb, int nb, int kb) {
  int lane = threadIdx.x & 31;
  const __bf16* p = Bt + (size_t)(nb + (lane & 15)) * ldb + kb + (lane >> 4) * 8;
  BF16x16 u;
  u.h[0] = *(const v8bf*)p;
  u.h[1] = *(const v8bf*)(p + 16);
  return u.v;
}

__device__ __forceinline__ v8f wmma_bf16(v16bf a, v16bf b, v8f c) {
  return __builtin_amdgcn_wmma_f32_16x16x32_bf16(false, a, false, b, (short)0, c, false, false);
}

__device__ __forceinline__ float redmax16(float v) {
  #pragma unroll
  for (int m = 1; m < 16; m <<= 1) v = fmaxf(v, __shfl_xor(v, m, 32));
  return v;
}
__device__ __forceinline__ float redsum16(float v) {
  #pragma unroll
  for (int m = 1; m < 16; m <<= 1) v += __shfl_xor(v, m, 32);
  return v;
}

// 32 rows x 64 cols per wave: two A fragments share every B fragment.
__device__ __forceinline__ void gemm32x64(const __bf16* __restrict__ A, int lda,
                                          const __bf16* __restrict__ Bt, int ldb,
                                          int nb, int K, F8 acc[2][4]) {
  for (int kb = 0; kb < K; kb += 32) {
    v16bf a0 = load_a_frag(A, lda, kb);
    v16bf a1 = load_a_frag(A + (size_t)16 * lda, lda, kb);
    #pragma unroll
    for (int f = 0; f < 4; ++f) {
      v16bf b = load_b_frag(Bt, ldb, nb + f * 16, kb);
      acc[0][f].v = wmma_bf16(a0, b, acc[0][f].v);
      acc[1][f].v = wmma_bf16(a1, b, acc[1][f].v);
    }
  }
}

__device__ __forceinline__ void zero_acc2(F8 acc[2][4]) {
  #pragma unroll
  for (int s = 0; s < 2; ++s)
    #pragma unroll
    for (int f = 0; f < 4; ++f)
      #pragma unroll
      for (int i = 0; i < 8; ++i) acc[s][f].f[i] = 0.f;
}

// ---------------------------------------------------------------------------
// Pre-pass kernels
// ---------------------------------------------------------------------------
__global__ void cvt_bf16_kernel(const float* __restrict__ in, __bf16* __restrict__ out, int n) {
  int i = blockIdx.x * blockDim.x + threadIdx.x;
  if (i < n) out[i] = f2bf(in[i]);
}

// W[K][N] (row-major) -> Wt[N][K] bf16
__global__ void wtrans_kernel(const float* __restrict__ W, __bf16* __restrict__ Wt, int K, int N) {
  int i = blockIdx.x * blockDim.x + threadIdx.x;
  if (i < K * N) {
    int n = i / K, k = i % K;
    Wt[i] = f2bf(W[(size_t)k * N + n]);
  }
}

// ---------------------------------------------------------------------------
// ckv = x @ Wdown, then LayerNorm over R=64, store bf16 [BT][64]
// one wave per 32-row strip; 2x4 fragments cover 32 rows x N=64
// ---------------------------------------------------------------------------
__global__ __launch_bounds__(128) void ckv_ln_kernel(const __bf16* __restrict__ xbf,
                                                     const __bf16* __restrict__ WdT,
                                                     const float* __restrict__ g,
                                                     const float* __restrict__ beta,
                                                     __bf16* __restrict__ out) {
  int w = blockIdx.x * 4 + (threadIdx.x >> 5);
  int lane = threadIdx.x & 31, half = lane >> 4, nl = lane & 15;
  int rowBase = w * 32;
  const __bf16* A = xbf + (size_t)rowBase * Cc;

  F8 acc[2][4];
  zero_acc2(acc);
  gemm32x64(A, Cc, WdT, Cc, 0, Cc, acc);

  #pragma unroll
  for (int sub = 0; sub < 2; ++sub) {
    #pragma unroll
    for (int r = 0; r < 8; ++r) {
      float s = acc[sub][0].f[r] + acc[sub][1].f[r] + acc[sub][2].f[r] + acc[sub][3].f[r];
      float q = acc[sub][0].f[r] * acc[sub][0].f[r] + acc[sub][1].f[r] * acc[sub][1].f[r] +
                acc[sub][2].f[r] * acc[sub][2].f[r] + acc[sub][3].f[r] * acc[sub][3].f[r];
      s = redsum16(s);
      q = redsum16(q);
      float mean = s * (1.f / 64.f);
      float var  = q * (1.f / 64.f) - mean * mean;
      float rn   = rsqrtf(var + 1e-5f);
      int row = rowBase + sub * 16 + r + 8 * half;
      #pragma unroll
      for (int f = 0; f < 4; ++f) {
        int d = f * 16 + nl;
        float v = (acc[sub][f].f[r] - mean) * rn * g[d] + beta[d];
        out[(size_t)row * Rc + d] = f2bf(v);
      }
    }
  }
}

// RoPE rotation of a 16-row sub-tile (d, d+-32 pair in same lane of f, f^2)
__device__ __forceinline__ void rope_store(F8 a[4], int b, int h, int tb,
                                           int half, int nl, __bf16* __restrict__ dst) {
  float invA = __expf(-LOG_THETA * (float)nl * (1.f / 32.f));          // d%32 = nl
  float invB = __expf(-LOG_THETA * (float)(16 + nl) * (1.f / 32.f));   // d%32 = 16+nl
  #pragma unroll
  for (int r = 0; r < 8; ++r) {
    int t = tb + r + 8 * half;
    float s0, c0, s1, c1;
    sincosf((float)t * invA, &s0, &c0);
    sincosf((float)t * invB, &s1, &c1);
    float o0 = a[0].f[r] * c0 - a[2].f[r] * s0;
    float o1 = a[1].f[r] * c1 - a[3].f[r] * s1;
    float o2 = a[2].f[r] * c0 + a[0].f[r] * s0;
    float o3 = a[3].f[r] * c1 + a[1].f[r] * s1;
    size_t base = (((size_t)b * Hc + h) * Tc + t) * Dc;
    dst[base + 0 * 16 + nl] = f2bf(o0);
    dst[base + 1 * 16 + nl] = f2bf(o1);
    dst[base + 2 * 16 + nl] = f2bf(o2);
    dst[base + 3 * 16 + nl] = f2bf(o3);
  }
}

// ---------------------------------------------------------------------------
// Q = x @ Wq, fused RoPE, store bf16 Q[B][H][T][D]
// one wave per (32-row strip, head)
// ---------------------------------------------------------------------------
__global__ __launch_bounds__(128) void q_rope_kernel(const __bf16* __restrict__ xbf,
                                                     const __bf16* __restrict__ WqT,
                                                     __bf16* __restrict__ Qb) {
  int w = blockIdx.x * 4 + (threadIdx.x >> 5);
  int strip = w % (BTc / 32);
  int h = w / (BTc / 32);
  int lane = threadIdx.x & 31, half = lane >> 4, nl = lane & 15;
  int rowBase = strip * 32;
  const __bf16* A = xbf + (size_t)rowBase * Cc;

  F8 acc[2][4];
  zero_acc2(acc);
  gemm32x64(A, Cc, WqT, Cc, h * Dc, Cc, acc);

  int b  = rowBase / Tc;
  int tb = rowBase % Tc;
  rope_store(acc[0], b, h, tb,      half, nl, Qb);
  rope_store(acc[1], b, h, tb + 16, half, nl, Qb);
}

// ---------------------------------------------------------------------------
// kv = ckv_ln @ Wup: group g in [0,24): part=g/12 (0=K, 1=V), head=g%12.
// K gets RoPE and [B][H][T][D]; V stored transposed [B][H][D][T] so the
// attention PV B-fragment loads are contiguous along t.
// ---------------------------------------------------------------------------
__global__ __launch_bounds__(128) void kv_kernel(const __bf16* __restrict__ ckv,
                                                 const __bf16* __restrict__ WuT,
                                                 __bf16* __restrict__ Kb,
                                                 __bf16* __restrict__ Vt) {
  int w = blockIdx.x * 4 + (threadIdx.x >> 5);
  int strip = w % (BTc / 32);
  int g = w / (BTc / 32);        // 0..23
  int part = g / Hc, h = g % Hc;
  int lane = threadIdx.x & 31, half = lane >> 4, nl = lane & 15;
  int rowBase = strip * 32;
  const __bf16* A = ckv + (size_t)rowBase * Rc;

  F8 acc[2][4];
  zero_acc2(acc);
  gemm32x64(A, Rc, WuT, Rc, g * Dc, Rc, acc);

  int b  = rowBase / Tc;
  int tb = rowBase % Tc;

  if (part == 0) {  // K: RoPE + row-major store
    rope_store(acc[0], b, h, tb,      half, nl, Kb);
    rope_store(acc[1], b, h, tb + 16, half, nl, Kb);
  } else {          // V: transposed store, 8 contiguous t per lane -> b128 store
    #pragma unroll
    for (int sub = 0; sub < 2; ++sub) {
      #pragma unroll
      for (int f = 0; f < 4; ++f) {
        int d = f * 16 + nl;
        v8bf pack;
        #pragma unroll
        for (int r = 0; r < 8; ++r) pack[r] = f2bf(acc[sub][f].f[r]);
        size_t base = (((size_t)b * Hc + h) * Dc + d) * Tc + tb + sub * 16 + 8 * half;
        *(v8bf*)(Vt + base) = pack;
      }
    }
  }
}

// ---------------------------------------------------------------------------
// Flash attention, causal. One wave per (b, h, 16-query tile).
// Per 32-key block: 4 WMMA for S, online softmax, P staged in LDS as bf16,
// re-read as A fragment, 4 WMMA for O update. Next K tile prefetched and
// current V tile loaded early so softmax VALU hides the load latency.
// ---------------------------------------------------------------------------
__global__ __launch_bounds__(128) void attn_kernel(const __bf16* __restrict__ Qb,
                                                   const __bf16* __restrict__ Kb,
                                                   const __bf16* __restrict__ Vt,
                                                   __bf16* __restrict__ attn) {
  int w = blockIdx.x * 4 + (threadIdx.x >> 5);
  int qt = w % (Tc / 16);
  int h  = (w / (Tc / 16)) % Hc;
  int b  = w / ((Tc / 16) * Hc);
  int lane = threadIdx.x & 31, half = lane >> 4, nl = lane & 15;
  int qbase = qt * 16;

  const __bf16* Qp = Qb + (((size_t)b * Hc + h) * Tc) * Dc;
  const __bf16* Kp = Kb + (((size_t)b * Hc + h) * Tc) * Dc;
  const __bf16* Vp = Vt + (((size_t)b * Hc + h) * Dc) * Tc;

  v16bf qa0 = load_a_frag(Qp + (size_t)qbase * Dc, Dc, 0);
  v16bf qa1 = load_a_frag(Qp + (size_t)qbase * Dc, Dc, 32);

  F8 o[4];
  float mrow[8], lrow[8];
  #pragma unroll
  for (int f = 0; f < 4; ++f)
    #pragma unroll
    for (int i = 0; i < 8; ++i) o[f].f[i] = 0.f;
  #pragma unroll
  for (int i = 0; i < 8; ++i) { mrow[i] = -1e30f; lrow[i] = 0.f; }

  __shared__ __bf16 psh[4][16 * 32];
  __bf16* ps = psh[threadIdx.x >> 5];

  // prefetch first K tile
  v16bf kf0 = load_b_frag(Kp, Dc, 0,  0);
  v16bf kf1 = load_b_frag(Kp, Dc, 0,  32);
  v16bf kf2 = load_b_frag(Kp, Dc, 16, 0);
  v16bf kf3 = load_b_frag(Kp, Dc, 16, 32);

  for (int kb = 0; kb <= qbase; kb += 32) {
    // issue current V tile loads early (consumed after softmax)
    v16bf vf0 = load_b_frag(Vp, Tc, 0,  kb);
    v16bf vf1 = load_b_frag(Vp, Tc, 16, kb);
    v16bf vf2 = load_b_frag(Vp, Tc, 32, kb);
    v16bf vf3 = load_b_frag(Vp, Tc, 48, kb);

    F8 s0, s1;
    #pragma unroll
    for (int i = 0; i < 8; ++i) { s0.f[i] = 0.f; s1.f[i] = 0.f; }
    s0.v = wmma_bf16(qa0, kf0, s0.v);
    s0.v = wmma_bf16(qa1, kf1, s0.v);
    s1.v = wmma_bf16(qa0, kf2, s1.v);
    s1.v = wmma_bf16(qa1, kf3, s1.v);

    // prefetch next K tile (wave-uniform branch, EXEC stays all-1s)
    if (kb + 32 <= qbase) {
      kf0 = load_b_frag(Kp, Dc, kb + 32, 0);
      kf1 = load_b_frag(Kp, Dc, kb + 32, 32);
      kf2 = load_b_frag(Kp, Dc, kb + 48, 0);
      kf3 = load_b_frag(Kp, Dc, kb + 48, 32);
    }

    bool diag = (kb == qbase);   // only the diagonal tile needs masking

    #pragma unroll
    for (int r = 0; r < 8; ++r) {
      float a0 = s0.f[r] * 0.125f;  // D^-0.5
      float a1 = s1.f[r] * 0.125f;
      if (diag) {
        int qrow = qbase + r + 8 * half;
        if (kb + nl      > qrow) a0 = -1e30f;
        if (kb + 16 + nl > qrow) a1 = -1e30f;
      }
      float mx = redmax16(fmaxf(a0, a1));
      float mn = fmaxf(mrow[r], mx);
      float alpha = __expf(mrow[r] - mn);
      a0 = __expf(a0 - mn);
      a1 = __expf(a1 - mn);
      float rs = redsum16(a0 + a1);
      lrow[r] = lrow[r] * alpha + rs;
      mrow[r] = mn;
      o[0].f[r] *= alpha; o[1].f[r] *= alpha;
      o[2].f[r] *= alpha; o[3].f[r] *= alpha;
      int rr = r + 8 * half;
      ps[rr * 32 + nl]      = f2bf(a0);   // same-wave LDS: DS ops are in-order
      ps[rr * 32 + 16 + nl] = f2bf(a1);
    }

    v16bf pa = load_a_frag(ps, 32, 0);
    o[0].v = wmma_bf16(pa, vf0, o[0].v);
    o[1].v = wmma_bf16(pa, vf1, o[1].v);
    o[2].v = wmma_bf16(pa, vf2, o[2].v);
    o[3].v = wmma_bf16(pa, vf3, o[3].v);
  }

  #pragma unroll
  for (int r = 0; r < 8; ++r) {
    float inv = 1.f / lrow[r];
    int t = qbase + r + 8 * half;
    size_t rowo = ((size_t)b * Tc + t) * (Hc * Dc) + h * Dc;
    #pragma unroll
    for (int f = 0; f < 4; ++f)
      attn[rowo + f * 16 + nl] = f2bf(o[f].f[r] * inv);
  }
}

// ---------------------------------------------------------------------------
// out = attn @ Wo + bo   (f32 output)
// ---------------------------------------------------------------------------
__global__ __launch_bounds__(128) void out_kernel(const __bf16* __restrict__ attn,
                                                  const __bf16* __restrict__ WoT,
                                                  const float* __restrict__ bo,
                                                  float* __restrict__ out) {
  int w = blockIdx.x * 4 + (threadIdx.x >> 5);
  int strip = w % (BTc / 32);
  int g = w / (BTc / 32);   // 0..11, col base g*64
  int lane = threadIdx.x & 31, half = lane >> 4, nl = lane & 15;
  int rowBase = strip * 32;
  const __bf16* A = attn + (size_t)rowBase * Cc;

  F8 acc[2][4];
  zero_acc2(acc);
  gemm32x64(A, Cc, WoT, Cc, g * Dc, Cc, acc);

  #pragma unroll
  for (int sub = 0; sub < 2; ++sub) {
    #pragma unroll
    for (int r = 0; r < 8; ++r) {
      int row = rowBase + sub * 16 + r + 8 * half;
      #pragma unroll
      for (int f = 0; f < 4; ++f) {
        int col = g * Dc + f * 16 + nl;
        out[(size_t)row * Cc + col] = acc[sub][f].f[r] + bo[col];
      }
    }
  }
}

// ---------------------------------------------------------------------------
// Launch
// ---------------------------------------------------------------------------
extern "C" void kernel_launch(void* const* d_in, const int* in_sizes, int n_in,
                              void* d_out, int out_size, void* d_ws, size_t ws_size,
                              hipStream_t stream) {
  const float* x     = (const float*)d_in[0];
  const float* Wq    = (const float*)d_in[1];
  const float* Wdown = (const float*)d_in[2];
  const float* ln_g  = (const float*)d_in[3];
  const float* ln_b  = (const float*)d_in[4];
  const float* Wup   = (const float*)d_in[5];
  const float* Wo    = (const float*)d_in[6];
  const float* bo    = (const float*)d_in[7];

  char* p = (char*)d_ws;
  auto take = [&](size_t bytes) -> void* {
    void* q = (void*)p;
    p += (bytes + 255) & ~(size_t)255;
    return q;
  };
  __bf16* xbf  = (__bf16*)take((size_t)BTc * Cc * 2);
  __bf16* WqT  = (__bf16*)take((size_t)Cc * Cc * 2);
  __bf16* WdT  = (__bf16*)take((size_t)Rc * Cc * 2);
  __bf16* WuT  = (__bf16*)take((size_t)(2 * Hc * Dc) * Rc * 2);
  __bf16* WoT  = (__bf16*)take((size_t)Cc * Cc * 2);
  __bf16* ckv  = (__bf16*)take((size_t)BTc * Rc * 2);
  __bf16* Qb   = (__bf16*)take((size_t)BTc * Cc * 2);
  __bf16* Kb   = (__bf16*)take((size_t)BTc * Cc * 2);
  __bf16* Vt   = (__bf16*)take((size_t)BTc * Cc * 2);
  __bf16* attn = (__bf16*)take((size_t)BTc * Cc * 2);

  const int nx = BTc * Cc;
  cvt_bf16_kernel<<<(nx + 255) / 256, 256, 0, stream>>>(x, xbf, nx);
  wtrans_kernel<<<(Cc * Cc + 255) / 256, 256, 0, stream>>>(Wq, WqT, Cc, Cc);
  wtrans_kernel<<<(Cc * Rc + 255) / 256, 256, 0, stream>>>(Wdown, WdT, Cc, Rc);
  wtrans_kernel<<<(Rc * 2 * Hc * Dc + 255) / 256, 256, 0, stream>>>(Wup, WuT, Rc, 2 * Hc * Dc);
  wtrans_kernel<<<(Cc * Cc + 255) / 256, 256, 0, stream>>>(Wo, WoT, Cc, Cc);

  ckv_ln_kernel<<<(BTc / 32) / 4, 128, 0, stream>>>(xbf, WdT, ln_g, ln_b, ckv);
  q_rope_kernel<<<((BTc / 32) * Hc) / 4, 128, 0, stream>>>(xbf, WqT, Qb);
  kv_kernel<<<((BTc / 32) * 2 * Hc) / 4, 128, 0, stream>>>(ckv, WuT, Kb, Vt);
  attn_kernel<<<(Bc * Hc * (Tc / 16)) / 4, 128, 0, stream>>>(Qb, Kb, Vt, attn);
  out_kernel<<<((BTc / 32) * Hc) / 4, 128, 0, stream>>>(attn, WoT, bo, (float*)d_out);
}
